// HeteroGAE_Decoder_48661979464093
// MI455X (gfx1250) — matrix operand order repaired and checked by reading, not compiled, not run on hardware
//
#include <hip/hip_runtime.h>
#include <hip/hip_bf16.h>

typedef __attribute__((ext_vector_type(16))) _Float16 v16h;
typedef __attribute__((ext_vector_type(8)))  _Float16 v8h;
typedef __attribute__((ext_vector_type(4)))  _Float16 v4h;
typedef __attribute__((ext_vector_type(8)))  float    v8f;

#define C_IN 128   // node feature width for conv stages

// ---------------------------------------------------------------------------
// zero fill
// ---------------------------------------------------------------------------
__global__ __launch_bounds__(256) void zero_f32(float* __restrict__ p, long n) {
  long i = (long)blockIdx.x * blockDim.x + threadIdx.x;
  if (i < n) p[i] = 0.0f;
}

// ---------------------------------------------------------------------------
// f32 -> f16 convert (for z)
// ---------------------------------------------------------------------------
__global__ __launch_bounds__(256) void f32_to_f16(const float* __restrict__ in,
                                                  _Float16* __restrict__ out,
                                                  long n) {
  long i = (long)blockIdx.x * blockDim.x + threadIdx.x;
  if (i < n) out[i] = (_Float16)in[i];
}

// ---------------------------------------------------------------------------
// weight prep: W f32 [K,Kout] row-major  ->  WT f16 [Kout,K] row-major
// (makes every WMMA B-fragment a contiguous 128-bit load)
// ---------------------------------------------------------------------------
__global__ __launch_bounds__(256) void transpose_w_f16(
    const float* __restrict__ W, _Float16* __restrict__ WT, int K, int Kout) {
  int t = blockIdx.x * blockDim.x + threadIdx.x;
  if (t >= K * Kout) return;
  int k = t / Kout, n = t % Kout;
  WT[(long)n * K + k] = (_Float16)W[t];
}

// ---------------------------------------------------------------------------
// scatter-sum + degree count for SAGE mean aggregation (f16 activations,
// f32 accumulation via global_atomic_add_f32). One wave32 per edge.
// ---------------------------------------------------------------------------
__global__ __launch_bounds__(256) void scatter_accum(
    const _Float16* __restrict__ x, const int* __restrict__ src,
    const int* __restrict__ dst, float* __restrict__ sum,
    float* __restrict__ cnt, int nEdges) {
  long t = (long)blockIdx.x * blockDim.x + threadIdx.x;
  int e = (int)(t >> 5);
  int lane = (int)(t & 31);
  if (e >= nEdges) return;
  int s = src[e], d = dst[e];
  v4h v = *(const v4h*)(x + (long)s * C_IN + lane * 4);
  float* o = sum + (long)d * C_IN + lane * 4;
  atomicAdd(o + 0, (float)v[0]);
  atomicAdd(o + 1, (float)v[1]);
  atomicAdd(o + 2, (float)v[2]);
  atomicAdd(o + 3, (float)v[3]);
  if (lane == 0) atomicAdd(cnt + d, 1.0f);
}

// ---------------------------------------------------------------------------
// one f16 GEMM term: acc += A[16 rows] @ WT^T, K in steps of 32.
// A-fragment: two contiguous b128 loads; B-fragment: two contiguous b128
// loads from the pre-transposed f16 weights.
// ---------------------------------------------------------------------------
__device__ __forceinline__ void wmma_term_f16(
    const _Float16* __restrict__ A, const _Float16* __restrict__ WT, int K,
    long mload, int kg, int colBase, int Kout, v8f acc[4]) {
  for (int kk = 0; kk < K; kk += 32) {
    const _Float16* ap = A + mload * K + kk;
    v8h lo = *(const v8h*)(ap + kg * 8);
    v8h hi = *(const v8h*)(ap + 16 + kg * 8);
    v16h a = __builtin_shufflevector(lo, hi, 0, 1, 2, 3, 4, 5, 6, 7, 8, 9, 10,
                                     11, 12, 13, 14, 15);
#pragma unroll
    for (int c = 0; c < 4; ++c) {
      int n = colBase + c * 16;
      int ncl = n < Kout ? n : Kout - 1;  // clamp; invalid cols never stored
      const _Float16* wp = WT + (long)ncl * K + kk + kg * 8;
      v8h b0 = *(const v8h*)(wp);
      v8h b1 = *(const v8h*)(wp + 16);
      v16h b = __builtin_shufflevector(b0, b1, 0, 1, 2, 3, 4, 5, 6, 7, 8, 9,
                                       10, 11, 12, 13, 14, 15);
      acc[c] = __builtin_amdgcn_wmma_f32_16x16x32_f16(
          false, a, false, b, (short)0, acc[c], false, false);
    }
  }
}

// ---------------------------------------------------------------------------
// WMMA GEMM:
//   out(f16) = act( [mean(Am)*1/deg @ Wm] + A1 @ W1 + [A2 @ W2] + bias )
// All weights pre-transposed f16 [Kout,K]. Am is the f32 atomic-sum buffer
// with the 1/max(cnt,1) scaling fused into the A-fragment converts.
// Each wave: 16 rows x 64 cols (4 f32 accumulators), v_wmma_f32_16x16x32_f16.
// ---------------------------------------------------------------------------
__global__ __launch_bounds__(256) void wmma_gemm(
    const float* __restrict__ Am, const _Float16* __restrict__ WmT,
    const float* __restrict__ cnt, const _Float16* __restrict__ A1,
    const _Float16* __restrict__ W1T, int K1, const _Float16* __restrict__ A2,
    const _Float16* __restrict__ W2T, int K2, const float* __restrict__ bias,
    _Float16* __restrict__ out, int Nrows, int Kout, int doRelu) {
  const int lane = threadIdx.x & 31;
  const int wave = threadIdx.x >> 5;
  const int nRowTiles = (Nrows + 15) >> 4;
  const int rowTile = blockIdx.x * 8 + wave;
  if (rowTile >= nRowTiles) return;  // wave-uniform exit

  const int row = lane & 15;  // M (A) / N (B,D) sub-index
  const int kg = lane >> 4;   // lane-group 0/1
  const int m = rowTile * 16 + row;
  const long mload = (m < Nrows ? m : Nrows - 1);
  const int colBase = blockIdx.y * 64 + row;

  v8f acc[4] = {};

  // ---- mean term (f32 sums, fused 1/deg scale) ----
  if (Am) {
    float scale = 1.0f / fmaxf(cnt[mload], 1.0f);
    for (int kk = 0; kk < K1; kk += 32) {
      const float* ap = Am + mload * K1 + kk;
      float4 f0 = *(const float4*)(ap + kg * 8);
      float4 f1 = *(const float4*)(ap + kg * 8 + 4);
      float4 f2 = *(const float4*)(ap + 16 + kg * 8);
      float4 f3 = *(const float4*)(ap + 16 + kg * 8 + 4);
      v16h a;
      a[0] = (_Float16)(f0.x * scale);  a[1] = (_Float16)(f0.y * scale);
      a[2] = (_Float16)(f0.z * scale);  a[3] = (_Float16)(f0.w * scale);
      a[4] = (_Float16)(f1.x * scale);  a[5] = (_Float16)(f1.y * scale);
      a[6] = (_Float16)(f1.z * scale);  a[7] = (_Float16)(f1.w * scale);
      a[8] = (_Float16)(f2.x * scale);  a[9] = (_Float16)(f2.y * scale);
      a[10] = (_Float16)(f2.z * scale); a[11] = (_Float16)(f2.w * scale);
      a[12] = (_Float16)(f3.x * scale); a[13] = (_Float16)(f3.y * scale);
      a[14] = (_Float16)(f3.z * scale); a[15] = (_Float16)(f3.w * scale);
#pragma unroll
      for (int c = 0; c < 4; ++c) {
        int n = colBase + c * 16;
        int ncl = n < Kout ? n : Kout - 1;
        const _Float16* wp = WmT + (long)ncl * K1 + kk + kg * 8;
        v8h b0 = *(const v8h*)(wp);
        v8h b1 = *(const v8h*)(wp + 16);
        v16h b = __builtin_shufflevector(b0, b1, 0, 1, 2, 3, 4, 5, 6, 7, 8, 9,
                                         10, 11, 12, 13, 14, 15);
        acc[c] = __builtin_amdgcn_wmma_f32_16x16x32_f16(
            false, a, false, b, (short)0, acc[c], false, false);
      }
    }
  }

  // ---- main f16 term(s) ----
  wmma_term_f16(A1, W1T, K1, mload, kg, colBase, Kout, acc);
  if (A2) wmma_term_f16(A2, W2T, K2, mload, kg, colBase, Kout, acc);

  // ---- epilogue: bias + optional relu; f32 D layout: VGPR r -> M = r+kg*8
#pragma unroll
  for (int c = 0; c < 4; ++c) {
    int n = colBase + c * 16;
    if (n < Kout) {
      float bv = bias ? bias[n] : 0.0f;
#pragma unroll
      for (int r = 0; r < 8; ++r) {
        int mm = rowTile * 16 + kg * 8 + r;
        if (mm < Nrows) {
          float v = acc[c][r] + bv;
          if (doRelu) v = fmaxf(v, 0.0f);
          out[(long)mm * Kout + n] = (_Float16)v;
        }
      }
    }
  }
}

// ---------------------------------------------------------------------------
// row-wise log_softmax over 20 f16 logits -> f32 d_out[0 : N*20]
// ---------------------------------------------------------------------------
__global__ __launch_bounds__(256) void logsoftmax20(
    const _Float16* __restrict__ logits, float* __restrict__ out, int N) {
  int i = blockIdx.x * blockDim.x + threadIdx.x;
  if (i >= N) return;
  const _Float16* p = logits + (long)i * 20;
  float v[20];
#pragma unroll
  for (int j = 0; j < 20; ++j) v[j] = (float)p[j];
  float mx = v[0];
#pragma unroll
  for (int j = 1; j < 20; ++j) mx = fmaxf(mx, v[j]);
  float s = 0.0f;
#pragma unroll
  for (int j = 0; j < 20; ++j) s += __expf(v[j] - mx);
  float l = __logf(s);
#pragma unroll
  for (int j = 0; j < 20; ++j) out[(long)i * 20 + j] = v[j] - mx - l;
}

// ---------------------------------------------------------------------------
// edge_probs = sigmoid(dot(h[src], h[dst])) ; one wave32 per edge,
// 4 halves per lane + shuffle butterfly reduction.
// ---------------------------------------------------------------------------
__global__ __launch_bounds__(256) void edge_dot_sigmoid(
    const _Float16* __restrict__ h, const int* __restrict__ ei, int E,
    float* __restrict__ out) {
  long t = (long)blockIdx.x * blockDim.x + threadIdx.x;
  int e = (int)(t >> 5);
  int lane = (int)(t & 31);
  if (e >= E) return;
  int a = ei[e], b = ei[(long)E + e];
  v4h x = *(const v4h*)(h + (long)a * C_IN + lane * 4);
  v4h y = *(const v4h*)(h + (long)b * C_IN + lane * 4);
  float d = (float)x[0] * (float)y[0] + (float)x[1] * (float)y[1] +
            (float)x[2] * (float)y[2] + (float)x[3] * (float)y[3];
#pragma unroll
  for (int off = 16; off > 0; off >>= 1) d += __shfl_xor(d, off, 32);
  if (lane == 0) out[e] = 1.0f / (1.0f + __expf(-d));
}

// ---------------------------------------------------------------------------
// host-side launch
// ---------------------------------------------------------------------------
static inline void launch_gemm(const float* Am, const _Float16* WmT,
                               const float* cnt, const _Float16* A1,
                               const _Float16* W1T, int K1, const _Float16* A2,
                               const _Float16* W2T, int K2, const float* bias,
                               _Float16* out, int N, int Kout, int relu,
                               hipStream_t s) {
  int rowTiles = (N + 15) / 16;
  dim3 grid((rowTiles + 7) / 8, (Kout + 63) / 64);
  wmma_gemm<<<grid, 256, 0, s>>>(Am, WmT, cnt, A1, W1T, K1, A2, W2T, K2, bias,
                                 out, N, Kout, relu);
}

static inline void launch_tw(const float* W, _Float16* WT, int K, int Kout,
                             hipStream_t s) {
  int n = K * Kout;
  transpose_w_f16<<<(n + 255) / 256, 256, 0, s>>>(W, WT, K, Kout);
}

extern "C" void kernel_launch(void* const* d_in, const int* in_sizes, int n_in,
                              void* d_out, int out_size, void* d_ws,
                              size_t ws_size, hipStream_t stream) {
  const float* z = (const float*)d_in[0];
  const int* ei = (const int*)d_in[1];  // [2,E]
  const int* bb = (const int*)d_in[2];  // [2,EB]
  const int N = in_sizes[0] / C_IN;
  const int E = in_sizes[1] / 2;
  const int EB = in_sizes[2] / 2;

  // params in setup_inputs() dict insertion order
  const float* convWl[3] = {(const float*)d_in[3], (const float*)d_in[6],
                            (const float*)d_in[9]};
  const float* convbl[3] = {(const float*)d_in[4], (const float*)d_in[7],
                            (const float*)d_in[10]};
  const float* convWr[3] = {(const float*)d_in[5], (const float*)d_in[8],
                            (const float*)d_in[11]};
  const float* linW = (const float*)d_in[12];
  const float* linB = (const float*)d_in[13];
  const float* decW[5] = {(const float*)d_in[14], (const float*)d_in[16],
                          (const float*)d_in[18], (const float*)d_in[20],
                          (const float*)d_in[22]};
  const float* decB[5] = {(const float*)d_in[15], (const float*)d_in[17],
                          (const float*)d_in[19], (const float*)d_in[21],
                          (const float*)d_in[23]};

  // ---- workspace carve-up: f32 region first, then f16 region
  float* meanbuf = (float*)d_ws;                    // N*128 f32
  float* cntbuf = meanbuf + (size_t)N * C_IN;       // N f32 (contiguous)
  _Float16* zh = (_Float16*)(cntbuf + N);           // N*128 f16
  _Float16* hA = zh + (size_t)N * C_IN;             // N*128
  _Float16* hB = hA + (size_t)N * C_IN;             // N*128
  _Float16* dA = hB + (size_t)N * C_IN;             // N*256
  _Float16* dB = dA + (size_t)N * 256;              // N*256
  _Float16* wts = dB + (size_t)N * 256;             // transposed f16 weights
  _Float16* wlT[3], *wrT[3];
  _Float16* p = wts;
  for (int l = 0; l < 3; ++l) { wlT[l] = p; p += 128 * 128; }
  for (int l = 0; l < 3; ++l) { wrT[l] = p; p += 128 * 128; }
  _Float16* linT = p;   p += 128 * 128;
  _Float16* d0aT = p;   p += 256 * 128;
  _Float16* d0bT = p;   p += 256 * 128;
  _Float16* d1T = p;    p += 256 * 256;
  _Float16* d2T = p;    p += 256 * 256;
  _Float16* d3T = p;    p += 256 * 256;
  _Float16* d4T = p;    p += 20 * 256;
  _Float16* logits = (_Float16*)meanbuf;  // reuse after convs (N*20 f16)

  float* o_xr = (float*)d_out;            // [N,20] f32
  float* o_edge = o_xr + (size_t)N * 20;  // [E] f32

  const int* bb_src = bb;       // row 0
  const int* bb_dst = bb + EB;  // row 1

  // ---- weight prep (tiny, L2-resident afterwards)
  for (int l = 0; l < 3; ++l) {
    launch_tw(convWl[l], wlT[l], 128, 128, stream);
    launch_tw(convWr[l], wrT[l], 128, 128, stream);
  }
  launch_tw(linW, linT, 128, 128, stream);
  launch_tw(decW[0], d0aT, 128, 256, stream);                   // rows 0..127
  launch_tw(decW[0] + (size_t)128 * 256, d0bT, 128, 256, stream);  // rows 128..255
  launch_tw(decW[1], d1T, 256, 256, stream);
  launch_tw(decW[2], d2T, 256, 256, stream);
  launch_tw(decW[3], d3T, 256, 256, stream);
  launch_tw(decW[4], d4T, 256, 20, stream);

  // ---- z -> f16
  long zn = (long)N * C_IN;
  f32_to_f16<<<(int)((zn + 255) / 256), 256, 0, stream>>>(z, zh, zn);

  // ---- 3x SAGEConv (zh -> hA -> hB -> hA): scatter-mean + fused dual GEMM
  const _Float16* cur = zh;
  _Float16* nxt = hA;
  for (int l = 0; l < 3; ++l) {
    long zlen = (long)N * (C_IN + 1);
    zero_f32<<<(int)((zlen + 255) / 256), 256, 0, stream>>>(meanbuf, zlen);
    long st = (long)EB * 32;
    scatter_accum<<<(int)((st + 255) / 256), 256, 0, stream>>>(
        cur, bb_src, bb_dst, meanbuf, cntbuf, EB);
    launch_gemm(meanbuf, wlT[l], cntbuf, cur, wrT[l], C_IN, nullptr, nullptr,
                0, convbl[l], nxt, N, C_IN, 1, stream);
    cur = nxt;
    nxt = (nxt == hA) ? hB : hA;
  }
  // cur == hA after 3 layers

  // ---- lin: h = hA @ linW + linB  (no relu) -> hB
  launch_gemm(nullptr, nullptr, nullptr, hA, linT, C_IN, nullptr, nullptr, 0,
              linB, hB, N, C_IN, 0, stream);
  const _Float16* h = hB;

  // ---- decoder: concat(z,h) via split dec0 weights
  launch_gemm(nullptr, nullptr, nullptr, zh, d0aT, C_IN, h, d0bT, C_IN,
              decB[0], dA, N, 256, 1, stream);
  launch_gemm(nullptr, nullptr, nullptr, dA, d1T, 256, nullptr, nullptr, 0,
              decB[1], dB, N, 256, 1, stream);
  launch_gemm(nullptr, nullptr, nullptr, dB, d2T, 256, nullptr, nullptr, 0,
              decB[2], dA, N, 256, 1, stream);
  launch_gemm(nullptr, nullptr, nullptr, dA, d3T, 256, nullptr, nullptr, 0,
              decB[3], dB, N, 256, 1, stream);
  launch_gemm(nullptr, nullptr, nullptr, dB, d4T, 256, nullptr, nullptr, 0,
              decB[4], logits, N, 20, 0, stream);

  logsoftmax20<<<(N + 255) / 256, 256, 0, stream>>>(logits, o_xr, N);

  long et = (long)E * 32;
  edge_dot_sigmoid<<<(int)((et + 255) / 256), 256, 0, stream>>>(h, ei, E,
                                                                o_edge);
}
